// DTCClusterLayer_76046690943287
// MI455X (gfx1250) — matrix operand order repaired and checked by reading, not compiled, not run on hardware
//
#include <hip/hip_runtime.h>
#include <stdint.h>

// Problem constants (from reference): N=65536, D_LAT=1024, K=256, ALPHA=1
#define D_LAT   1024
#define KC      256
#define EPS_D   1e-12f
#define ROWS    16          // z rows per block in the GEMM kernel
#define KSTAGE  512         // K staged in LDS per phase (2 phases cover 1024)

typedef unsigned short ushort_t;
typedef __attribute__((ext_vector_type(16))) __bf16   v16bf;
typedef __attribute__((ext_vector_type(8)))  float    v8f;
typedef __attribute__((ext_vector_type(8)))  ushort_t v8u;
typedef __attribute__((ext_vector_type(16))) ushort_t v16u;

// ---- bf16 helpers (round-to-nearest-even split: x ~= hi + lo) ----
__device__ __forceinline__ ushort_t f2bf(float x) {
  unsigned u = __float_as_uint(x);
  u += 0x7FFFu + ((u >> 16) & 1u);
  return (ushort_t)(u >> 16);
}
__device__ __forceinline__ float bf2f(ushort_t h) {
  return __uint_as_float(((unsigned)h) << 16);
}

// Build a 16-element bf16 fragment from two 8-element (16-byte) vector loads.
__device__ __forceinline__ v16bf frag16(const ushort_t* p, const ushort_t* q) {
  v8u a = *(const v8u*)p;
  v8u b = *(const v8u*)q;
  v16u u = __builtin_shufflevector(a, b, 0,1,2,3,4,5,6,7,8,9,10,11,12,13,14,15);
  return __builtin_bit_cast(v16bf, u);
}

__device__ __forceinline__ v8f wmma_bf16(v16bf a, v16bf b, v8f c) {
  // (neg_a, A, neg_b, B, c_mod, C, reuse_a, reuse_b)
  return __builtin_amdgcn_wmma_f32_16x16x32_bf16(false, a, false, b, (short)0, c,
                                                 false, false);
}

// ---------------------------------------------------------------------------
// Kernel 1: split centroids [256 x 1024] f32 -> bf16 hi/lo, compute ||c||^2.
// ---------------------------------------------------------------------------
__global__ __launch_bounds__(256) void centConv(const float* __restrict__ cent,
                                                ushort_t* __restrict__ chi,
                                                ushort_t* __restrict__ clo,
                                                float* __restrict__ c2) {
  const int r = blockIdx.x;      // centroid row
  const int t = threadIdx.x;
  float ssq = 0.f;
#pragma unroll
  for (int j = 0; j < D_LAT / 256; ++j) {
    int col = t + (j << 8);
    float v = cent[(size_t)r * D_LAT + col];
    ssq += v * v;
    ushort_t h = f2bf(v);
    ushort_t l = f2bf(v - bf2f(h));
    chi[(size_t)r * D_LAT + col] = h;
    clo[(size_t)r * D_LAT + col] = l;
  }
  // wave reduce then cross-wave via LDS (fixed order -> deterministic)
  ssq += __shfl_xor(ssq, 1);  ssq += __shfl_xor(ssq, 2);
  ssq += __shfl_xor(ssq, 4);  ssq += __shfl_xor(ssq, 8);
  ssq += __shfl_xor(ssq, 16);
  __shared__ float s[8];
  if ((t & 31) == 0) s[t >> 5] = ssq;
  __syncthreads();
  if (t == 0) {
    float a = 0.f;
#pragma unroll
    for (int w = 0; w < 8; ++w) a += s[w];
    c2[r] = a;
  }
}

__global__ void zeroF(float* __restrict__ F) { F[threadIdx.x] = 0.f; }

// ---------------------------------------------------------------------------
// Kernel 2: fused z-copy + bf16x3 WMMA GEMM + Student-t + row-normalize (Q)
//           + per-block column partial sums of Q.
// Block: 256 threads = 8 waves; covers 16 z rows x all 256 centroids.
// Wave w handles column tiles n0 = 32w and 32w+16 (16x16 WMMA tiles).
// ---------------------------------------------------------------------------
__global__ __launch_bounds__(256) void dtc_gemm_q(
    const float* __restrict__ z, const ushort_t* __restrict__ chi,
    const ushort_t* __restrict__ clo, const float* __restrict__ c2,
    float* __restrict__ outZ, float* __restrict__ outQ,
    float* __restrict__ fdst, int useAtomic) {
  __shared__ ushort_t zhi[ROWS * KSTAGE];   // 16 KB
  __shared__ ushort_t zlo[ROWS * KSTAGE];   // 16 KB
  __shared__ float z2row[ROWS];
  __shared__ float rspart[8][ROWS];
  __shared__ float rowsum[ROWS];

  const int t    = threadIdx.x;
  const int row0 = blockIdx.x * ROWS;
  const int r    = t >> 4;                  // load-phase row 0..15
  const int c0   = t & 15;                  // load-phase col group
  const int wv   = t >> 5;
  const int lane = t & 31;
  const int half = lane >> 4;               // wave32 half (WMMA layout)
  const int lr   = lane & 15;
  const int n0a  = wv * 32;
  const int n0b  = n0a + 16;

  v8f acc0 = {};
  v8f acc1 = {};
  float ssq = 0.f;

  for (int stage = 0; stage < D_LAT / KSTAGE; ++stage) {
    const int ks = stage * KSTAGE;
    // -- stream 16 x KSTAGE f32 z values once: copy to out, split to LDS --
#pragma unroll 4
    for (int j = 0; j < KSTAGE / 16; ++j) {
      int col = c0 + (j << 4);
      size_t gi = (size_t)(row0 + r) * D_LAT + ks + col;
      float v = z[gi];
      outZ[gi] = v;                          // fused z -> output copy
      ssq += v * v;
      ushort_t h = f2bf(v);
      ushort_t l = f2bf(v - bf2f(h));
      zhi[r * KSTAGE + col] = h;
      zlo[r * KSTAGE + col] = l;
    }
    __syncthreads();

    // -- bf16x3 WMMA K-loop over this stage --
    for (int kk = 0; kk < KSTAGE; kk += 32) {
      // A fragment (16x32 bf16): lane half0 -> K kk..kk+15,
      // half1 -> K kk+16..kk+31  (contiguous, matches B)
      const ushort_t* ap = &zhi[lr * KSTAGE + kk + half * 16];
      const ushort_t* lp = &zlo[lr * KSTAGE + kk + half * 16];
      v16bf ahi = frag16(ap, ap + 8);
      v16bf alo = frag16(lp, lp + 8);

      // B fragment (32x16 bf16): lane = column; half0 -> K kk..kk+15,
      // half1 -> K kk+16..kk+31. centroids row-major => contiguous loads.
      const int kg = ks + kk + half * 16;
      const ushort_t* b0h = &chi[(size_t)(n0a + lr) * D_LAT + kg];
      const ushort_t* b0l = &clo[(size_t)(n0a + lr) * D_LAT + kg];
      const ushort_t* b1h = &chi[(size_t)(n0b + lr) * D_LAT + kg];
      const ushort_t* b1l = &clo[(size_t)(n0b + lr) * D_LAT + kg];
      v16bf bh0 = frag16(b0h, b0h + 8);
      v16bf bl0 = frag16(b0l, b0l + 8);
      v16bf bh1 = frag16(b1h, b1h + 8);
      v16bf bl1 = frag16(b1l, b1l + 8);

      // fp32-ish product: hi*hi + hi*lo + lo*hi
      acc0 = wmma_bf16(ahi, bh0, acc0);
      acc0 = wmma_bf16(ahi, bl0, acc0);
      acc0 = wmma_bf16(alo, bh0, acc0);
      acc1 = wmma_bf16(ahi, bh1, acc1);
      acc1 = wmma_bf16(ahi, bl1, acc1);
      acc1 = wmma_bf16(alo, bh1, acc1);
    }
    __syncthreads();   // before next stage overwrites LDS
  }

  // ||z_row||^2 : reduce across the 16 threads that loaded each row
  float s2 = ssq;
  s2 += __shfl_xor(s2, 1);  s2 += __shfl_xor(s2, 2);
  s2 += __shfl_xor(s2, 4);  s2 += __shfl_xor(s2, 8);
  if (c0 == 0) z2row[r] = s2;
  __syncthreads();

  // C/D layout: lane half0 holds rows m=0..7 (vgpr i), half1 rows m=8..15;
  // column n = lr + tile base.
  const float c2a = c2[n0a + lr];
  const float c2b = c2[n0b + lr];
  float num0[8], num1[8];
#pragma unroll
  for (int i = 0; i < 8; ++i) {
    float zz  = z2row[half * 8 + i];
    float d2a = zz + c2a - 2.f * acc0[i];
    float d2b = zz + c2b - 2.f * acc1[i];
    float Da  = sqrtf(fmaxf(d2a, EPS_D));
    float Db  = sqrtf(fmaxf(d2b, EPS_D));
    num0[i] = 1.f / (1.f + Da);     // (1+D)^-(alpha+1)/2 with alpha=1
    num1[i] = 1.f / (1.f + Db);
  }

  // deterministic row sums: per-wave partials, then fixed-order sum
#pragma unroll
  for (int i = 0; i < 8; ++i) {
    float p = num0[i] + num1[i];
    p += __shfl_xor(p, 1);  p += __shfl_xor(p, 2);
    p += __shfl_xor(p, 4);  p += __shfl_xor(p, 8);
    if (lr == 0) rspart[wv][half * 8 + i] = p;
  }
  __syncthreads();
  if (t < ROWS) {
    float a = 0.f;
#pragma unroll
    for (int w = 0; w < 8; ++w) a += rspart[w][t];
    rowsum[t] = a;
  }
  __syncthreads();

  float cs0 = 0.f, cs1 = 0.f;
#pragma unroll
  for (int i = 0; i < 8; ++i) {
    float inv = 1.f / rowsum[half * 8 + i];
    float q0 = num0[i] * inv;
    float q1 = num1[i] * inv;
    outQ[(size_t)(row0 + half * 8 + i) * KC + n0a + lr] = q0;
    outQ[(size_t)(row0 + half * 8 + i) * KC + n0b + lr] = q1;
    cs0 += q0;
    cs1 += q1;
  }
  // combine the two halves (same column n, different row groups)
  cs0 += __shfl_xor(cs0, 16);
  cs1 += __shfl_xor(cs1, 16);
  if (half == 0) {
    if (useAtomic) {   // fallback only if workspace too small for partials
      atomicAdd(&fdst[n0a + lr], cs0);
      atomicAdd(&fdst[n0b + lr], cs1);
    } else {           // per-block partial sums (deterministic path)
      fdst[(size_t)blockIdx.x * KC + n0a + lr] = cs0;
      fdst[(size_t)blockIdx.x * KC + n0b + lr] = cs1;
    }
  }
}

// ---------------------------------------------------------------------------
// Kernel 3: F[n] = sum over blocks of column partials (fixed-order tree).
// ---------------------------------------------------------------------------
__global__ __launch_bounds__(256) void reduceF(const float* __restrict__ pF,
                                               float* __restrict__ F, int nb) {
  __shared__ float s[256];
  const int t = threadIdx.x;
  const int n = blockIdx.x;   // column
  float a = 0.f;
  for (int b = t; b < nb; b += 256) a += pF[(size_t)b * KC + n];
  s[t] = a;
  __syncthreads();
  for (int off = 128; off > 0; off >>= 1) {
    if (t < off) s[t] += s[t + off];
    __syncthreads();
  }
  if (t == 0) F[n] = s[0];
}

// ---------------------------------------------------------------------------
// Kernel 4: P = rownorm(Q*Q/F). One wave per row (K=256 -> 8 per lane).
// ---------------------------------------------------------------------------
__global__ __launch_bounds__(256) void dtc_p(const float* __restrict__ Q,
                                             const float* __restrict__ F,
                                             float* __restrict__ P) {
  const int wv = threadIdx.x >> 5;
  const int lane = threadIdx.x & 31;
  const size_t row = (size_t)blockIdx.x * 8 + wv;
  const float* qr = Q + row * KC;
  float v[8];
  float s = 0.f;
#pragma unroll
  for (int j = 0; j < 8; ++j) {
    int n = lane + (j << 5);
    float q = qr[n];
    float np = q * q / F[n];
    v[j] = np;
    s += np;
  }
  s += __shfl_xor(s, 1);  s += __shfl_xor(s, 2);  s += __shfl_xor(s, 4);
  s += __shfl_xor(s, 8);  s += __shfl_xor(s, 16);
  float inv = 1.f / s;
  float* pr = P + row * KC;
#pragma unroll
  for (int j = 0; j < 8; ++j) pr[lane + (j << 5)] = v[j] * inv;
}

// ---------------------------------------------------------------------------
extern "C" void kernel_launch(void* const* d_in, const int* in_sizes, int n_in,
                              void* d_out, int out_size, void* d_ws,
                              size_t ws_size, hipStream_t stream) {
  const float* z    = (const float*)d_in[0];
  const float* cent = (const float*)d_in[1];
  const int N = in_sizes[0] / D_LAT;         // 65536
  const int nblocks = N / ROWS;              // 4096

  float* out  = (float*)d_out;
  float* outZ = out;                                    // [N, 1024]
  float* outQ = out + (size_t)N * D_LAT;                // [N, 256]
  float* outP = outQ + (size_t)N * KC;                  // [N, 256]

  // workspace layout
  char* ws = (char*)d_ws;
  const size_t szHalf = (size_t)KC * D_LAT * sizeof(ushort_t);   // 512 KB
  ushort_t* chi = (ushort_t*)ws;
  ushort_t* clo = (ushort_t*)(ws + szHalf);
  float* c2 = (float*)(ws + 2 * szHalf);
  float* F  = c2 + KC;
  float* pF = F + KC;
  const size_t need = 2 * szHalf + 2 * KC * sizeof(float) +
                      (size_t)nblocks * KC * sizeof(float);       // ~5.2 MB
  const int useAtomic = (ws_size < need) ? 1 : 0;

  centConv<<<KC, 256, 0, stream>>>(cent, chi, clo, c2);
  if (useAtomic) zeroF<<<1, KC, 0, stream>>>(F);
  dtc_gemm_q<<<nblocks, 256, 0, stream>>>(z, chi, clo, c2, outZ, outQ,
                                          useAtomic ? F : pF, useAtomic);
  if (!useAtomic) reduceF<<<KC, 256, 0, stream>>>(pF, F, nblocks);
  dtc_p<<<N / 8, 256, 0, stream>>>(outQ, F, outP);
}
